// SelfAttentionLayer_26697516712217
// MI455X (gfx1250) — matrix operand order repaired and checked by reading, not compiled, not run on hardware
//
#include <hip/hip_runtime.h>

typedef _Float16 half_t;
typedef __attribute__((ext_vector_type(16))) _Float16 v16h;
typedef __attribute__((ext_vector_type(8)))  _Float16 v8h;
typedef __attribute__((ext_vector_type(8)))  float    v8f;

namespace {
constexpr int kB = 16, kS = 12, kN = 325, kD = 256, kF = 2048, kH = 8, kHD = 32;
constexpr int kM  = kB * kS * kN;   // 62400 tokens (divisible by 16)
constexpr int kBS = kB * kS;        // 192
constexpr int kMT = kM / 16;        // 3900 row tiles
constexpr int NP = 336;             // node dim padded to 21*16
constexpr int SP = 352;             // score K dim padded to 22*16 (mult of 32)
constexpr float kEps   = 1e-5f;
constexpr float kScale = 0.17677669529663687f;  // 1/sqrt(32)
}

union FragH { v16h v; v8h h[2]; };

// A fragment (16x32 f16): lane holds row M=lane%16, 16 contiguous K starting at
// k0 + 16*(lane>=16). Two 16B loads.
__device__ __forceinline__ v16h frag_a(const half_t* base, int ldk, int row, int k0, int lane) {
  const half_t* p = base + (size_t)row * (size_t)ldk + k0 + 16 * (lane >> 4);
  FragH f; f.h[0] = *(const v8h*)p; f.h[1] = *(const v8h*)(p + 8); return f.v;
}
// B fragment (32x16 f16) from BT[col][K] layout:
// lane holds column col, 16 contiguous K starting at k0 + 16*(lane>=16).
__device__ __forceinline__ v16h frag_b(const half_t* base, int ldk, int col, int k0, int lane) {
  const half_t* p = base + (size_t)col * (size_t)ldk + k0 + 16 * (lane >> 4);
  FragH f; f.h[0] = *(const v8h*)p; f.h[1] = *(const v8h*)(p + 8); return f.v;
}
__device__ __forceinline__ v8f wmma16(v16h a, v16h b, v8f c) {
  return __builtin_amdgcn_wmma_f32_16x16x32_f16(false, a, false, b, (short)0, c, false, false);
}

__global__ __launch_bounds__(256) void cvt_f32_to_f16(const float* __restrict__ src,
                                                      half_t* __restrict__ dst, size_t n) {
  for (size_t i = blockIdx.x * 256ull + threadIdx.x; i < n; i += (size_t)gridDim.x * 256ull)
    dst[i] = (half_t)src[i];
}

// W [K][Nc] f32 -> WT [Nc][K] f16
__global__ __launch_bounds__(256) void transpose_to_f16(const float* __restrict__ W,
                                                        half_t* __restrict__ WT, int K, int Nc) {
  size_t n = (size_t)K * Nc;
  for (size_t i = blockIdx.x * 256ull + threadIdx.x; i < n; i += (size_t)gridDim.x * 256ull) {
    int k = (int)(i / Nc), c = (int)(i % Nc);
    WT[(size_t)c * K + k] = (half_t)W[i];
  }
}

// per-head scatter for q/k: out[(bs*H+h)*N + n][hd]
__device__ __forceinline__ void scatter_qk(half_t* __restrict__ out, const v8f& c, float bv,
                                           int tm, int tn, int lo, int hi) {
  int f = tn * 16 + lo, h = f >> 5, d = f & 31;
  for (int r = 0; r < 8; r++) {
    int m = tm * 16 + r + 8 * hi;
    int bs = m / kN, n = m % kN;
    out[((size_t)(bs * kH + h) * kN + n) * kHD + d] = (half_t)(c[r] + bv);
  }
}
// per-head transposed scatter for v: out[(bs*H+h)*hd + d][N]
__device__ __forceinline__ void scatter_vT(half_t* __restrict__ out, const v8f& c, float bv,
                                           int tm, int tn, int lo, int hi) {
  int f = tn * 16 + lo, h = f >> 5, d = f & 31;
  for (int r = 0; r < 8; r++) {
    int m = tm * 16 + r + 8 * hi;
    int bs = m / kN, n = m % kN;
    out[((size_t)(bs * kH + h) * kHD + d) * kN + n] = (half_t)(c[r] + bv);
  }
}

// Fused QKV: one wave owns one 16-row strip; A fragments hoisted to registers once
// and reused across 3 weight matrices x 16 N-tiles (48x A-traffic reduction).
__global__ __launch_bounds__(256) void qkv_fused(const half_t* __restrict__ x16,
                                                 const half_t* __restrict__ WqT,
                                                 const half_t* __restrict__ WkT,
                                                 const half_t* __restrict__ WvT,
                                                 const float* __restrict__ bq,
                                                 const float* __restrict__ bk,
                                                 const float* __restrict__ bv,
                                                 half_t* __restrict__ q16,
                                                 half_t* __restrict__ k16,
                                                 half_t* __restrict__ vT16) {
  int lane = threadIdx.x & 31, wave = threadIdx.x >> 5;
  int lo = lane & 15, hi = lane >> 4;
  int tm = blockIdx.x * 8 + wave;
  if (tm >= kMT) return;
  v16h aF[8];
#pragma unroll
  for (int kk = 0; kk < 8; kk++) aF[kk] = frag_a(x16, kD, tm * 16 + lo, kk * 32, lane);

  for (int tn = 0; tn < 16; tn++) {
    v8f c = {};
#pragma unroll
    for (int kk = 0; kk < 8; kk++) c = wmma16(aF[kk], frag_b(WqT, kD, tn * 16 + lo, kk * 32, lane), c);
    scatter_qk(q16, c, bq[tn * 16 + lo], tm, tn, lo, hi);
  }
  for (int tn = 0; tn < 16; tn++) {
    v8f c = {};
#pragma unroll
    for (int kk = 0; kk < 8; kk++) c = wmma16(aF[kk], frag_b(WkT, kD, tn * 16 + lo, kk * 32, lane), c);
    scatter_qk(k16, c, bk[tn * 16 + lo], tm, tn, lo, hi);
  }
  for (int tn = 0; tn < 16; tn++) {
    v8f c = {};
#pragma unroll
    for (int kk = 0; kk < 8; kk++) c = wmma16(aF[kk], frag_b(WvT, kD, tn * 16 + lo, kk * 32, lane), c);
    scatter_vT(vT16, c, bv[tn * 16 + lo], tm, tn, lo, hi);
  }
}

// Fully fused attention for one (b,s,h): scores + softmax + attn@V, all in LDS.
__global__ __launch_bounds__(256) void attention_fused(const half_t* __restrict__ q16,
                                                       const half_t* __restrict__ k16,
                                                       const half_t* __restrict__ vT16,
                                                       half_t* __restrict__ ao16) {
  extern __shared__ char smem_raw[];
  half_t* sQ  = (half_t*)smem_raw;     // [NP][32]
  half_t* sK  = sQ  + NP * 32;         // [NP][32]
  half_t* sVT = sK  + NP * 32;         // [32][SP]
  half_t* sS  = sVT + 32 * SP;         // [NP][SP]
  int tid = threadIdx.x;
  int lane = tid & 31, wave = tid >> 5, lo = lane & 15, hi = lane >> 4;
  int bh = blockIdx.x;                 // bs*H + h
  int bs = bh >> 3, h = bh & 7;
  const half_t* qg = q16 + (size_t)bh * kN * kHD;
  const half_t* kg = k16 + (size_t)bh * kN * kHD;
  const half_t* vg = vT16 + (size_t)bh * kHD * kN;

  for (int i = tid; i < NP * 32; i += 256) {
    int n = i >> 5, d = i & 31;
    half_t qv = (half_t)0, kv = (half_t)0;
    if (n < kN) { qv = qg[n * kHD + d]; kv = kg[n * kHD + d]; }
    sQ[i] = qv; sK[i] = kv;
  }
  for (int i = tid; i < 32 * SP; i += 256) {
    int d = i / SP, n = i % SP;
    sVT[i] = (n < kN) ? vg[d * kN + n] : (half_t)0;
  }
  for (int i = tid; i < NP * 16; i += 256) {   // zero pad K-tile cols 336..351 of scores
    int r = i >> 4, cc = 336 + (i & 15);
    sS[(size_t)r * SP + cc] = (half_t)0;
  }
  __syncthreads();

  // scores = Q K^T * scale ; 21x21 tiles, hd=32 -> one WMMA per tile
  for (int t = wave; t < 21 * 21; t += 8) {
    int tm = t / 21, tn = t % 21;
    v16h a = frag_a(sQ, 32, tm * 16 + lo, 0, lane);
    v16h b = frag_b(sK, 32, tn * 16 + lo, 0, lane);
    v8f z = {};
    v8f c = wmma16(a, b, z);
    int col = tn * 16 + lo;
    for (int r = 0; r < 8; r++)
      sS[(size_t)(tm * 16 + r + 8 * hi) * SP + col] = (half_t)(c[r] * kScale);
  }
  __syncthreads();

  // row softmax over valid columns 0..324 (f32 math, f16 storage)
  for (int r = tid; r < NP; r += 256) {
    half_t* row = sS + (size_t)r * SP;
    if (r < kN) {
      float mx = -3.0e38f;
      for (int m = 0; m < kN; m++) { float v = (float)row[m]; mx = v > mx ? v : mx; }
      float sum = 0.f;
      for (int m = 0; m < kN; m++) { float e = __expf((float)row[m] - mx); sum += e; row[m] = (half_t)e; }
      float inv = 1.f / sum;
      for (int m = 0; m < kN; m++) row[m] = (half_t)((float)row[m] * inv);
      for (int m = kN; m < NP; m++) row[m] = (half_t)0;
    } else {
      for (int m = 0; m < NP; m++) row[m] = (half_t)0;
    }
  }
  __syncthreads();

  // out = attn @ V ; 21 x 2 tiles, K = 352 -> 11 WMMA steps; A loaded once per
  // K-step feeding both head-dim accumulators.
  for (int tn = wave; tn < 21; tn += 8) {
    v8f c0 = {}, c1 = {};
    for (int kk = 0; kk < SP; kk += 32) {
      v16h a  = frag_a(sS,  SP, tn * 16 + lo, kk, lane);
      v16h b0 = frag_b(sVT, SP, lo,      kk, lane);
      v16h b1 = frag_b(sVT, SP, 16 + lo, kk, lane);
      c0 = wmma16(a, b0, c0);
      c1 = wmma16(a, b1, c1);
    }
    for (int r = 0; r < 8; r++) {
      int n = tn * 16 + r + 8 * hi;
      if (n < kN) {
        size_t rowbase = ((size_t)bs * kN + n) * kD + h * kHD;
        ao16[rowbase + lo]      = (half_t)c0[r];
        ao16[rowbase + 16 + lo] = (half_t)c1[r];
      }
    }
  }
}

// out-proj GEMM + bias + residual + LayerNorm for one 16-token tile.
// A fragments hoisted once, two accumulators per wave.
__global__ __launch_bounds__(256) void proj_ln(const half_t* __restrict__ ao16,
                                               const half_t* __restrict__ WoT,
                                               const float* __restrict__ bo,
                                               const float* __restrict__ xres,
                                               const float* __restrict__ g1,
                                               const float* __restrict__ be1,
                                               float* __restrict__ y1f,
                                               half_t* __restrict__ y1h) {
  __shared__ float sOut[16][kD];
  __shared__ float sMu[16], sRstd[16];
  int tid = threadIdx.x, lane = tid & 31, wave = tid >> 5, lo = lane & 15, hi = lane >> 4;
  int m0 = blockIdx.x * 16;
  v16h aF[8];
#pragma unroll
  for (int kk = 0; kk < 8; kk++) aF[kk] = frag_a(ao16, kD, m0 + lo, kk * 32, lane);
  int tn0 = wave * 2, tn1 = wave * 2 + 1;
  v8f c0 = {}, c1 = {};
#pragma unroll
  for (int kk = 0; kk < 8; kk++) {
    c0 = wmma16(aF[kk], frag_b(WoT, kD, tn0 * 16 + lo, kk * 32, lane), c0);
    c1 = wmma16(aF[kk], frag_b(WoT, kD, tn1 * 16 + lo, kk * 32, lane), c1);
  }
  {
    int col0 = tn0 * 16 + lo, col1 = tn1 * 16 + lo;
    float b0 = bo[col0], b1 = bo[col1];
    for (int r = 0; r < 8; r++) {
      sOut[r + 8 * hi][col0] = c0[r] + b0;
      sOut[r + 8 * hi][col1] = c1[r] + b1;
    }
  }
  __syncthreads();
  for (int i = tid; i < 16 * kD; i += 256) {
    int t = i >> 8, j = i & 255;
    sOut[t][j] += xres[(size_t)(m0 + t) * kD + j];
  }
  __syncthreads();
  if (tid < 16) {
    float mu = 0.f;
    for (int j = 0; j < kD; j++) mu += sOut[tid][j];
    mu *= (1.f / kD);
    float var = 0.f;
    for (int j = 0; j < kD; j++) { float d = sOut[tid][j] - mu; var += d * d; }
    var *= (1.f / kD);
    sMu[tid] = mu; sRstd[tid] = rsqrtf(var + kEps);
  }
  __syncthreads();
  for (int i = tid; i < 16 * kD; i += 256) {
    int t = i >> 8, j = i & 255;
    float y = (sOut[t][j] - sMu[t]) * sRstd[t] * g1[j] + be1[j];
    size_t idx = (size_t)(m0 + t) * kD + j;
    y1f[idx] = y;
    y1h[idx] = (half_t)y;
  }
}

// FFN1 + ReLU (hidden in LDS) + FFN2 + bias + residual + LayerNorm per 16-token tile.
__global__ __launch_bounds__(256) void ffn_fused(const half_t* __restrict__ y1h,
                                                 const half_t* __restrict__ W1T,
                                                 const float* __restrict__ b1,
                                                 const half_t* __restrict__ W2T,
                                                 const float* __restrict__ b2,
                                                 const float* __restrict__ y1f,
                                                 const float* __restrict__ g2,
                                                 const float* __restrict__ be2,
                                                 float* __restrict__ out) {
  extern __shared__ char smem_raw2[];
  half_t* sH    = (half_t*)smem_raw2;                             // [16][kF] f16
  float*  sO    = (float*)(smem_raw2 + 16 * kF * 2);              // [16][kD] f32
  float*  sMu   = (float*)(smem_raw2 + 16 * kF * 2 + 16 * kD * 4);
  float*  sRstd = sMu + 16;
  int tid = threadIdx.x, lane = tid & 31, wave = tid >> 5, lo = lane & 15, hi = lane >> 4;
  int m0 = blockIdx.x * 16;
  // FFN1: [16,256] @ [256,2048] -> sH (ReLU). A hoisted once, dual accumulators.
  v16h aF[8];
#pragma unroll
  for (int kk = 0; kk < 8; kk++) aF[kk] = frag_a(y1h, kD, m0 + lo, kk * 32, lane);
  for (int i = 0; i < 8; i++) {
    int tn0 = wave * 16 + 2 * i, tn1 = tn0 + 1;
    v8f c0 = {}, c1 = {};
#pragma unroll
    for (int kk = 0; kk < 8; kk++) {
      c0 = wmma16(aF[kk], frag_b(W1T, kD, tn0 * 16 + lo, kk * 32, lane), c0);
      c1 = wmma16(aF[kk], frag_b(W1T, kD, tn1 * 16 + lo, kk * 32, lane), c1);
    }
    int col0 = tn0 * 16 + lo, col1 = tn1 * 16 + lo;
    float bb0 = b1[col0], bb1 = b1[col1];
    for (int r = 0; r < 8; r++) {
      float v0 = c0[r] + bb0, v1 = c1[r] + bb1;
      size_t rb = (size_t)(r + 8 * hi) * kF;
      sH[rb + col0] = (half_t)(v0 > 0.f ? v0 : 0.f);
      sH[rb + col1] = (half_t)(v1 > 0.f ? v1 : 0.f);
    }
  }
  __syncthreads();
  // FFN2: [16,2048] @ [2048,256] -> sO. One A load feeds two accumulators.
  {
    int tn0 = wave * 2, tn1 = wave * 2 + 1;
    v8f c0 = {}, c1 = {};
    for (int k = 0; k < kF; k += 32) {
      v16h a = frag_a(sH, kF, lo, k, lane);
      c0 = wmma16(a, frag_b(W2T, kF, tn0 * 16 + lo, k, lane), c0);
      c1 = wmma16(a, frag_b(W2T, kF, tn1 * 16 + lo, k, lane), c1);
    }
    int col0 = tn0 * 16 + lo, col1 = tn1 * 16 + lo;
    float bb0 = b2[col0], bb1 = b2[col1];
    for (int r = 0; r < 8; r++) {
      size_t rb = (size_t)(r + 8 * hi) * kD;
      sO[rb + col0] = c0[r] + bb0;
      sO[rb + col1] = c1[r] + bb1;
    }
  }
  __syncthreads();
  for (int i = tid; i < 16 * kD; i += 256) {
    int t = i >> 8, j = i & 255;
    sO[(size_t)t * kD + j] += y1f[(size_t)(m0 + t) * kD + j];
  }
  __syncthreads();
  if (tid < 16) {
    float mu = 0.f;
    for (int j = 0; j < kD; j++) mu += sO[tid * kD + j];
    mu *= (1.f / kD);
    float var = 0.f;
    for (int j = 0; j < kD; j++) { float d = sO[tid * kD + j] - mu; var += d * d; }
    var *= (1.f / kD);
    sMu[tid] = mu; sRstd[tid] = rsqrtf(var + kEps);
  }
  __syncthreads();
  for (int i = tid; i < 16 * kD; i += 256) {
    int t = i >> 8, j = i & 255;
    out[(size_t)(m0 + t) * kD + j] = (sO[(size_t)t * kD + j] - sMu[t]) * sRstd[t] * g2[j] + be2[j];
  }
}

extern "C" void kernel_launch(void* const* d_in, const int* in_sizes, int n_in,
                              void* d_out, int out_size, void* d_ws, size_t ws_size,
                              hipStream_t stream) {
  (void)in_sizes; (void)n_in; (void)out_size; (void)ws_size;
  const float* x    = (const float*)d_in[0];
  const float* Wq   = (const float*)d_in[1];
  const float* bq   = (const float*)d_in[2];
  const float* Wk   = (const float*)d_in[3];
  const float* bk   = (const float*)d_in[4];
  const float* Wv   = (const float*)d_in[5];
  const float* bv   = (const float*)d_in[6];
  const float* Wo   = (const float*)d_in[7];
  const float* bo   = (const float*)d_in[8];
  const float* W1   = (const float*)d_in[9];
  const float* b1   = (const float*)d_in[10];
  const float* W2   = (const float*)d_in[11];
  const float* b2   = (const float*)d_in[12];
  const float* g1   = (const float*)d_in[13];
  const float* be1  = (const float*)d_in[14];
  const float* g2   = (const float*)d_in[15];
  const float* be2  = (const float*)d_in[16];
  // d_in[17] = dim (always -2 -> identity transpose)
  float* out = (float*)d_out;

  // workspace carve-out (bump allocator, 256B aligned)
  char* ws = (char*)d_ws;
  size_t off = 0;
  auto alloc = [&](size_t bytes) -> char* {
    char* p = ws + off;
    off += (bytes + 255) & ~(size_t)255;
    return p;
  };
  half_t* x16  = (half_t*)alloc((size_t)kM * kD * 2);
  half_t* WqT  = (half_t*)alloc((size_t)kD * kD * 2);
  half_t* WkT  = (half_t*)alloc((size_t)kD * kD * 2);
  half_t* WvT  = (half_t*)alloc((size_t)kD * kD * 2);
  half_t* WoT  = (half_t*)alloc((size_t)kD * kD * 2);
  half_t* W1T  = (half_t*)alloc((size_t)kF * kD * 2);   // [F][D]
  half_t* W2T  = (half_t*)alloc((size_t)kD * kF * 2);   // [D][F]
  half_t* q16  = (half_t*)alloc((size_t)kBS * kH * kN * kHD * 2);
  half_t* k16  = (half_t*)alloc((size_t)kBS * kH * kN * kHD * 2);
  half_t* vT16 = (half_t*)alloc((size_t)kBS * kH * kN * kHD * 2);
  half_t* ao16 = (half_t*)alloc((size_t)kM * kD * 2);
  float*  y1f  = (float*)alloc((size_t)kM * kD * 4);
  half_t* y1h  = (half_t*)alloc((size_t)kM * kD * 2);

  const int attnSmem = (NP * 32 * 2 + 32 * SP + NP * SP) * 2;  // 302080 B
  const int ffnSmem  = 16 * kF * 2 + 16 * kD * 4 + 32 * 4;     // 82048 B
  hipFuncSetAttribute(reinterpret_cast<const void*>(attention_fused),
                      hipFuncAttributeMaxDynamicSharedMemorySize, attnSmem);
  hipFuncSetAttribute(reinterpret_cast<const void*>(ffn_fused),
                      hipFuncAttributeMaxDynamicSharedMemorySize, ffnSmem);

  // 1. precision conversion / weight transposition
  cvt_f32_to_f16<<<4096, 256, 0, stream>>>(x, x16, (size_t)kM * kD);
  transpose_to_f16<<<512,  256, 0, stream>>>(Wq, WqT, kD, kD);
  transpose_to_f16<<<512,  256, 0, stream>>>(Wk, WkT, kD, kD);
  transpose_to_f16<<<512,  256, 0, stream>>>(Wv, WvT, kD, kD);
  transpose_to_f16<<<512,  256, 0, stream>>>(Wo, WoT, kD, kD);
  transpose_to_f16<<<2048, 256, 0, stream>>>(W1, W1T, kD, kF);
  transpose_to_f16<<<2048, 256, 0, stream>>>(W2, W2T, kF, kD);

  // 2. fused QKV projection (A fragments register-resident, reused 48x)
  qkv_fused<<<(kMT + 7) / 8, 256, 0, stream>>>(x16, WqT, WkT, WvT, bq, bk, bv, q16, k16, vT16);

  // 3. fused attention per (b,s,h): 1536 workgroups, ~295 KB LDS each
  attention_fused<<<kBS * kH, 256, attnSmem, stream>>>(q16, k16, vT16, ao16);

  // 4. output projection + residual + LayerNorm 1
  proj_ln<<<kMT, 256, 0, stream>>>(ao16, WoT, bo, x, g1, be1, y1f, y1h);

  // 5. fused FFN + residual + LayerNorm 2 -> final f32 output
  ffn_fused<<<kMT, 256, ffnSmem, stream>>>(y1h, W1T, b1, W2T, b2, y1f, g2, be2, out);
}
